// Model_33225867002316
// MI455X (gfx1250) — compile-verified, matrix operand order checked by reading, not run on hardware
//
#include <hip/hip_runtime.h>
#include <hip/hip_bf16.h>
#include <math.h>

// ---------------------------------------------------------------------------
// BSRNN speech enhancer for MI455X (gfx1250, wave32, WMMA).
// All GEMMs use v_wmma_f32_16x16x32_f16 with fragment-pre-swizzled f16
// weights (one 32B v16h load per B fragment) and fragment-ordered LDS A
// tiles (one 32B v16h ds load per A fragment). Packed weights are padded to
// whole 64-column super-tiles so the GEMM inner loop is branch-free.
// ---------------------------------------------------------------------------

typedef __attribute__((ext_vector_type(16))) _Float16 v16h;
typedef __attribute__((ext_vector_type(8)))  float    v8f;

#define BATCH   2
#define LSAMP   32000
#define NFFT    512
#define HOPSZ   128
#define TFRM    251            // 1 + (32000+512-512)/128
#define FBIN    257
#define NBAND   31
#define FEATC   128
#define HID     256
#define G4      1024           // 4*HID
#define PADLEN  32512          // NFFT + (TFRM-1)*HOP

__device__ __forceinline__ float hannf(int i) {
    return 0.5f - 0.5f * cosf(6.283185307179586f * (float)i * (1.0f / 512.0f));
}
__device__ __forceinline__ float sigf(float x) { return 1.0f / (1.0f + expf(-x)); }

// K index inside a 32-wide K tile for the 16-bit WMMA A/B fragment layout
// (ISA 7.12.2): e<8 -> kk = e + 8*half ; e>=8 -> kk = 16 + (e-8) + 8*half.
__device__ __forceinline__ int wmma_kmap(int e, int half) {
    return (e < 8) ? (e + 8 * half) : (16 + (e - 8) + 8 * half);
}
// Inverse: kk (0..31) -> (half, e)
__device__ __forceinline__ int kk_half(int kk) { return (kk >> 3) & 1; }
__device__ __forceinline__ int kk_e(int kk)    { return (kk & 7) + ((kk & 16) ? 8 : 0); }

// ---------------------------------------------------------------------------
// Pack W[N][K] (f32) into WMMA-B fragment order (f16), zero-padded to
// n16pad 16-col tiles (n16pad = multiple of 4 so GEMM needs no tile guard):
// P[(((n16*k32)+kt)*32 + lane)*16 + e] = W[n16*16 + (lane&15)][kt*32 + kmap(e,lane>>4)]
// ---------------------------------------------------------------------------
__global__ void pack_frag_kernel(const float* __restrict__ W, _Float16* __restrict__ P,
                                 int N, int K, int n16pad) {
    int k32 = K >> 5;
    int total = n16pad * k32 * 512;
    int idx = blockIdx.x * blockDim.x + threadIdx.x;
    if (idx >= total) return;
    int e = idx & 15, lane = (idx >> 4) & 31;
    int rest = idx >> 9;
    int kt = rest % k32, nt = rest / k32;
    int row = nt * 16 + (lane & 15);
    int kk = kt * 32 + wmma_kmap(e, lane >> 4);
    P[idx] = (row < N) ? (_Float16)W[(size_t)row * K + kk] : (_Float16)0.f;
}

// ---------------------------------------------------------------------------
// STFT: one block per (b, frame). Reflect pad 256, Hann window, direct DFT.
// Output layout: sre/sim [B][257][T].
// ---------------------------------------------------------------------------
__global__ void stft_kernel(const float* __restrict__ x,
                            float* __restrict__ sre, float* __restrict__ sim) {
    int bt = blockIdx.x; int b = bt / TFRM; int t = bt % TFRM;
    __shared__ float fr[NFFT];
    for (int s = threadIdx.x; s < NFFT; s += blockDim.x) {
        int src = t * HOPSZ + s - NFFT / 2;
        if (src < 0) src = -src;
        if (src >= LSAMP) src = 2 * LSAMP - 2 - src;
        fr[s] = x[(size_t)b * LSAMP + src] * hannf(s);
    }
    __syncthreads();
    for (int k = threadIdx.x; k < FBIN; k += blockDim.x) {
        float ar = 0.f, ai = 0.f;
        float w = -6.283185307179586f * (float)k / 512.0f;
        for (int n = 0; n < NFFT; ++n) {
            float sn, cn; sincosf(w * (float)n, &sn, &cn);
            ar += fr[n] * cn; ai += fr[n] * sn;
        }
        sre[((size_t)b * FBIN + k) * TFRM + t] = ar;
        sim[((size_t)b * FBIN + k) * TFRM + t] = ai;
    }
}

// ---------------------------------------------------------------------------
// Band split + GroupNorm + 1x1 conv for one band. grid = BATCH blocks.
// ---------------------------------------------------------------------------
__global__ void bandsplit_kernel(const float* __restrict__ sre,
                                 const float* __restrict__ sim,
                                 int f0, int bw, int bandIdx,
                                 const float* __restrict__ gw, const float* __restrict__ gb,
                                 const float* __restrict__ cw, const float* __restrict__ cb,
                                 float* __restrict__ h) {
    int b = blockIdx.x;
    int ic = 2 * bw;                       // <= 32
    __shared__ float xs[32 * TFRM];
    __shared__ float r1[256], r2[256];
    float sum = 0.f, sq = 0.f;
    int n = ic * TFRM;
    for (int i = threadIdx.x; i < n; i += blockDim.x) {
        int c = i / TFRM, t = i % TFRM;
        float v = (c < bw)
            ? sre[((size_t)b * FBIN + f0 + c) * TFRM + t]
            : sim[((size_t)b * FBIN + f0 + c - bw) * TFRM + t];
        xs[i] = v; sum += v; sq += v * v;
    }
    r1[threadIdx.x] = sum; r2[threadIdx.x] = sq;
    __syncthreads();
    for (int s = 128; s > 0; s >>= 1) {
        if (threadIdx.x < s) { r1[threadIdx.x] += r1[threadIdx.x + s]; r2[threadIdx.x] += r2[threadIdx.x + s]; }
        __syncthreads();
    }
    float mu = r1[0] / (float)n;
    float var = r2[0] / (float)n - mu * mu;
    float rs = rsqrtf(var + 1e-6f);
    for (int i = threadIdx.x; i < n; i += blockDim.x) {
        int c = i / TFRM;
        xs[i] = (xs[i] - mu) * rs * gw[c] + gb[c];
    }
    __syncthreads();
    for (int idx = threadIdx.x; idx < FEATC * TFRM; idx += blockDim.x) {
        int f = idx / TFRM, t = idx % TFRM;
        float acc = cb[f];
        for (int c = 0; c < ic; ++c) acc += xs[c * TFRM + t] * cw[f * ic + c];
        h[(((size_t)b * NBAND + bandIdx) * FEATC + f) * TFRM + t] = acc;
    }
}

// ---------------------------------------------------------------------------
// GroupNorm over (C,Tq) per sample + transpose to [s][t][c]. grid = S blocks.
// ---------------------------------------------------------------------------
__global__ void gn_transpose_kernel(const float* __restrict__ x, size_t samp_stride,
                                    int C, int Tq,
                                    const float* __restrict__ gw, const float* __restrict__ gb,
                                    float* __restrict__ y) {
    int s = blockIdx.x;
    const float* xs = x + (size_t)s * samp_stride;
    __shared__ float r1[256], r2[256];
    int n = C * Tq;
    float sum = 0.f, sq = 0.f;
    for (int i = threadIdx.x; i < n; i += blockDim.x) { float v = xs[i]; sum += v; sq += v * v; }
    r1[threadIdx.x] = sum; r2[threadIdx.x] = sq;
    __syncthreads();
    for (int st = 128; st > 0; st >>= 1) {
        if (threadIdx.x < st) { r1[threadIdx.x] += r1[threadIdx.x + st]; r2[threadIdx.x] += r2[threadIdx.x + st]; }
        __syncthreads();
    }
    float mu = r1[0] / (float)n;
    float var = r2[0] / (float)n - mu * mu;
    float rs = rsqrtf(var + 1e-6f);
    for (int i = threadIdx.x; i < n; i += blockDim.x) {
        int c = i / Tq, t = i % Tq;
        y[((size_t)s * Tq + t) * C + c] = (xs[i] - mu) * rs * gw[c] + gb[c];
    }
}

__global__ void zero_kernel(float* __restrict__ p, int n) {
    int i = blockIdx.x * blockDim.x + threadIdx.x;
    if (i < n) p[i] = 0.f;
}

// ---------------------------------------------------------------------------
// WMMA GEMM: C[M,N] = A[M,K](f32) * W^T (+bias1+bias2)[+tanh]
// Wp = fragment-packed f16 weights padded to multiple-of-4 16-col tiles,
// so the K loop is completely branch-free. Block 128 threads (4 waves),
// tile 64 rows x 64 cols, K step 32.
// ---------------------------------------------------------------------------
__global__ __launch_bounds__(128)
void wmma_gemm_kernel(const float* __restrict__ A, const _Float16* __restrict__ Wp,
                      const float* __restrict__ bias1, const float* __restrict__ bias2,
                      float* __restrict__ C, int M, int N, int K, int act) {
    __shared__ _Float16 As[4][32][16];            // fragment-ordered A tile (4 KB)
    int bm = blockIdx.y * 64, bn = blockIdx.x * 64;
    int wv = threadIdx.x >> 5, lane = threadIdx.x & 31;
    int m = lane & 15, half = lane >> 4;
    int nk32 = K >> 5;
    size_t tstride = (size_t)nk32 * 512;          // halves per 16-col tile in Wp
    int bn16 = bn >> 4;
    v8f acc0 = {}, acc1 = {}, acc2 = {}, acc3 = {};
    for (int k0 = 0; k0 < K; k0 += 32) {
        // stage A k-tile into fragment-ordered LDS (vector loads, f32->f16)
        for (int i = threadIdx.x; i < 64 * 8; i += 128) {
            int r = i >> 3, c4 = (i & 7) << 2;    // 4 consecutive k
            int gr = bm + r;
            float4 v = {0.f, 0.f, 0.f, 0.f};
            if (gr < M) v = *(const float4*)(A + (size_t)gr * K + k0 + c4);
            _Float16* dst = &As[r >> 4][kk_half(c4) * 16 + (r & 15)][kk_e(c4)];
            dst[0] = (_Float16)v.x; dst[1] = (_Float16)v.y;
            dst[2] = (_Float16)v.z; dst[3] = (_Float16)v.w;
        }
        __syncthreads();
        v16h a = *(const v16h*)&As[wv][lane][0];
        const _Float16* bp = Wp + (((size_t)bn16 * nk32 + (k0 >> 5)) * 32 + lane) * 16;
        if (k0 + 32 < K)
            __builtin_prefetch((const void*)(bp + 512), 0, 0);   // next K tile
        v16h b0 = *(const v16h*)(bp);
        v16h b1 = *(const v16h*)(bp + tstride);
        v16h b2 = *(const v16h*)(bp + 2 * tstride);
        v16h b3 = *(const v16h*)(bp + 3 * tstride);
        acc0 = __builtin_amdgcn_wmma_f32_16x16x32_f16(false, a, false, b0, (short)0, acc0, false, false);
        acc1 = __builtin_amdgcn_wmma_f32_16x16x32_f16(false, a, false, b1, (short)0, acc1, false, false);
        acc2 = __builtin_amdgcn_wmma_f32_16x16x32_f16(false, a, false, b2, (short)0, acc2, false, false);
        acc3 = __builtin_amdgcn_wmma_f32_16x16x32_f16(false, a, false, b3, (short)0, acc3, false, false);
        __syncthreads();
    }
    v8f accs[4] = {acc0, acc1, acc2, acc3};
#pragma unroll
    for (int nt = 0; nt < 4; ++nt) {
        int col = bn + nt * 16 + m;
        if (col >= N) continue;
        float bv = (bias1 ? bias1[col] : 0.f) + (bias2 ? bias2[col] : 0.f);
#pragma unroll
        for (int r = 0; r < 8; ++r) {
            int row = bm + 16 * wv + r + 8 * half;
            if (row < M) {
                float v = accs[nt][r] + bv;
                if (act == 1) v = tanhf(v);
                C[(size_t)row * N + col] = v;
            }
        }
    }
}

// ---------------------------------------------------------------------------
// LSTM recurrence. One block (128 thr / 4 waves) owns 16 batch rows.
// xg: [S][Tq][1024] already includes bih+bhh. whhP: fragment-packed
// (N=1024,K=256). h state lives in LDS in fragment order so each WMMA A
// fragment is one contiguous 32B v16h load. Wave w computes gate quarter w.
// ---------------------------------------------------------------------------
__global__ __launch_bounds__(128)
void lstm_scan_kernel(const float* __restrict__ xg, const _Float16* __restrict__ whhP,
                      float* __restrict__ hcat, int S, int Tq, int hoff, int reverse) {
    __shared__ _Float16 hpk[8][32][16];    // fragment-ordered h (8 KB)
    __shared__ float    cst[16][HID];      // cell state (16 KB)
    __shared__ _Float16 gat[16][G4];       // gate pre-activations (32 KB)
    int wv = threadIdx.x >> 5, lane = threadIdx.x & 31;
    int m = lane & 15, half = lane >> 4;
    for (int i = threadIdx.x; i < 8 * 32 * 16; i += 128)
        ((_Float16*)hpk)[i] = (_Float16)0.f;
    for (int i = threadIdx.x; i < 16 * HID; i += 128)
        ((float*)cst)[i] = 0.f;
    __syncthreads();
    for (int step = 0; step < Tq; ++step) {
        int t = reverse ? (Tq - 1 - step) : step;
        // --- gates = h @ Whh^T (this wave's 256-col quarter) ---
        for (int nt = 0; nt < 16; ++nt) {
            int n16 = wv * 16 + nt;
            const _Float16* bp = whhP + (((size_t)n16 * 8) * 32 + lane) * 16;
            v8f acc = {};
#pragma unroll
            for (int kt = 0; kt < 8; ++kt) {
                v16h a = *(const v16h*)&hpk[kt][lane][0];
                v16h b = *(const v16h*)(bp + (size_t)kt * 512);
                acc = __builtin_amdgcn_wmma_f32_16x16x32_f16(
                    false, a, false, b, (short)0, acc, false, false);
            }
            int ncol = wv * 256 + nt * 16 + m;
#pragma unroll
            for (int r = 0; r < 8; ++r)
                gat[r + 8 * half][ncol] = (_Float16)acc[r];
        }
        __syncthreads();
        // --- elementwise LSTM cell update ---
        for (int idx = threadIdx.x; idx < 16 * HID; idx += 128) {
            int row = idx >> 8, j = idx & 255;
            int gr = blockIdx.x * 16 + row;
            bool valid = gr < S;
            const float* xgp = valid ? (xg + ((size_t)gr * Tq + t) * G4) : nullptr;
            float gi = (float)gat[row][j]           + (valid ? xgp[j] : 0.f);
            float gf = (float)gat[row][HID + j]     + (valid ? xgp[HID + j] : 0.f);
            float gg = (float)gat[row][2 * HID + j] + (valid ? xgp[2 * HID + j] : 0.f);
            float go = (float)gat[row][3 * HID + j] + (valid ? xgp[3 * HID + j] : 0.f);
            float c = sigf(gf) * cst[row][j] + sigf(gi) * tanhf(gg);
            float hh = sigf(go) * tanhf(c);
            cst[row][j] = c;
            int kt = j >> 5, kk = j & 31;
            hpk[kt][kk_half(kk) * 16 + row][kk_e(kk)] = (_Float16)hh;
            if (valid) hcat[((size_t)gr * Tq + t) * 512 + hoff + j] = hh;
        }
        __syncthreads();
    }
}

// ---------------------------------------------------------------------------
// Residual add with transpose: x[s][c][t] += po[s*Tq+t][c]
// ---------------------------------------------------------------------------
__global__ void add_transpose_kernel(float* __restrict__ x, const float* __restrict__ po,
                                     int S, int C, int Tq) {
    int idx = blockIdx.x * blockDim.x + threadIdx.x;
    int total = S * C * Tq;
    if (idx >= total) return;
    int t = idx % Tq, c = (idx / Tq) % C, s = idx / (Tq * C);
    x[idx] += po[((size_t)s * Tq + t) * C + c];
}

// h [b][(n*128+c)][t]  <->  xc [((b*T+t)*128+c)][n]
__global__ void gather_comm_kernel(const float* __restrict__ h, float* __restrict__ xc) {
    int idx = blockIdx.x * blockDim.x + threadIdx.x;
    int total = BATCH * TFRM * FEATC * NBAND;
    if (idx >= total) return;
    int n = idx % NBAND, c = (idx / NBAND) % FEATC;
    int t = (idx / (NBAND * FEATC)) % TFRM, b = idx / (NBAND * FEATC * TFRM);
    xc[idx] = h[(((size_t)b * NBAND + n) * FEATC + c) * TFRM + t];
}
__global__ void scatter_comm_kernel(float* __restrict__ h, const float* __restrict__ xc) {
    int idx = blockIdx.x * blockDim.x + threadIdx.x;
    int total = BATCH * TFRM * FEATC * NBAND;
    if (idx >= total) return;
    int n = idx % NBAND, c = (idx / NBAND) % FEATC;
    int t = (idx / (NBAND * FEATC)) % TFRM, b = idx / (NBAND * FEATC * TFRM);
    h[(((size_t)b * NBAND + n) * FEATC + c) * TFRM + t] = xc[idx];
}

// ---------------------------------------------------------------------------
// Mask application for one band: m3 [b*T+t][4*bw] -> est spec
// ---------------------------------------------------------------------------
__global__ void mask_apply_kernel(const float* __restrict__ m3,
                                  const float* __restrict__ sre, const float* __restrict__ sim,
                                  int f0, int bw,
                                  float* __restrict__ ere, float* __restrict__ eim) {
    int idx = blockIdx.x * blockDim.x + threadIdx.x;
    int total = BATCH * bw * TFRM;
    if (idx >= total) return;
    int t = idx % TFRM, k = (idx / TFRM) % bw, b = idx / (TFRM * bw);
    const float* mp = m3 + ((size_t)b * TFRM + t) * (4 * bw);
    float mr = mp[k]      * sigf(mp[2 * bw + k]);
    float mi = mp[bw + k] * sigf(mp[3 * bw + k]);
    size_t off = ((size_t)b * FBIN + f0 + k) * TFRM + t;
    float sr = sre[off], si = sim[off];
    ere[off] = sr * mr - si * mi;
    eim[off] = sr * mi + si * mr;
}

// ---------------------------------------------------------------------------
// iSTFT: irDFT per frame * Hann, overlap-add into padded buffer (atomics).
// ---------------------------------------------------------------------------
__global__ void istft_kernel(const float* __restrict__ ere, const float* __restrict__ eim,
                             float* __restrict__ padded) {
    int bt = blockIdx.x; int b = bt / TFRM; int t = bt % TFRM;
    __shared__ float sr[FBIN], si[FBIN];
    for (int i = threadIdx.x; i < FBIN; i += blockDim.x) {
        sr[i] = ere[((size_t)b * FBIN + i) * TFRM + t];
        si[i] = eim[((size_t)b * FBIN + i) * TFRM + t];
    }
    __syncthreads();
    for (int n = threadIdx.x; n < NFFT; n += blockDim.x) {
        float acc = sr[0] + ((n & 1) ? -sr[256] : sr[256]);
        float w = 6.283185307179586f * (float)n / 512.0f;
        for (int k = 1; k < 256; ++k) {
            float sn, cn; sincosf(w * (float)k, &sn, &cn);
            acc += 2.f * (sr[k] * cn - si[k] * sn);
        }
        float val = acc * (1.0f / 512.0f) * hannf(n);
        atomicAdd(&padded[(size_t)b * PADLEN + t * HOPSZ + n], val);
    }
}

__global__ void istft_norm_kernel(const float* __restrict__ padded, float* __restrict__ out) {
    int idx = blockIdx.x * blockDim.x + threadIdx.x;
    if (idx >= BATCH * LSAMP) return;
    int b = idx / LSAMP, l = idx % LSAMP;
    int p = l + NFFT / 2;
    int tmp = p - (NFFT - 1);
    int tlo = tmp <= 0 ? 0 : ((tmp + HOPSZ - 1) >> 7);
    int thi = p >> 7; if (thi > TFRM - 1) thi = TFRM - 1;
    float wsq = 0.f;
    for (int t = tlo; t <= thi; ++t) { float w = hannf(p - t * HOPSZ); wsq += w * w; }
    float v = padded[(size_t)b * PADLEN + p];
    out[idx] = v / (wsq > 1e-11f ? wsq : 1.f);
}

// ===========================================================================
// Host orchestration
// ===========================================================================
struct RESP {
    const float *gnw, *gnb, *wihf, *whhf, *bihf, *bhhf,
                *wihb, *whhb, *bihb, *bhhb, *projw, *projb;
};

extern "C" void kernel_launch(void* const* d_in, const int* in_sizes, int n_in,
                              void* d_out, int out_size, void* d_ws, size_t ws_size,
                              hipStream_t stream) {
    (void)in_sizes; (void)out_size; (void)ws_size;
    if (n_in < 469) return;

    // ---- unpack inputs (setup_inputs dict insertion order) ----
    int p = 0;
    const float* noisy = (const float*)d_in[p++];
    struct { const float *gnw, *gnb, *cw, *cb; } bn[NBAND];
    for (int i = 0; i < NBAND; ++i) {
        bn[i].gnw = (const float*)d_in[p++]; bn[i].gnb = (const float*)d_in[p++];
        bn[i].cw  = (const float*)d_in[p++]; bn[i].cb  = (const float*)d_in[p++];
    }
    RESP sep[4][2];
    for (int r = 0; r < 4; ++r)
        for (int part = 0; part < 2; ++part) {
            RESP& q = sep[r][part];
            q.gnw  = (const float*)d_in[p++]; q.gnb  = (const float*)d_in[p++];
            q.wihf = (const float*)d_in[p++]; q.whhf = (const float*)d_in[p++];
            q.bihf = (const float*)d_in[p++]; q.bhhf = (const float*)d_in[p++];
            q.wihb = (const float*)d_in[p++]; q.whhb = (const float*)d_in[p++];
            q.bihb = (const float*)d_in[p++]; q.bhhb = (const float*)d_in[p++];
            q.projw = (const float*)d_in[p++]; q.projb = (const float*)d_in[p++];
        }
    struct { const float *gnw, *gnb, *w1, *b1, *w2, *b2, *w3, *b3; } mk[NBAND];
    for (int i = 0; i < NBAND; ++i) {
        mk[i].gnw = (const float*)d_in[p++]; mk[i].gnb = (const float*)d_in[p++];
        mk[i].w1 = (const float*)d_in[p++];  mk[i].b1 = (const float*)d_in[p++];
        mk[i].w2 = (const float*)d_in[p++];  mk[i].b2 = (const float*)d_in[p++];
        mk[i].w3 = (const float*)d_in[p++];  mk[i].b3 = (const float*)d_in[p++];
    }

    // ---- band geometry ----
    int BW[NBAND], F0[NBAND];
    {
        int f = 0;
        for (int i = 0; i < NBAND; ++i) {
            BW[i] = (i < 10) ? 3 : (i < 22) ? 8 : (i < 30) ? 16 : 3;
            F0[i] = f; f += BW[i];
        }
    }

    // ---- workspace bump allocator ----
    char* wptr = (char*)d_ws;
    auto alloc = [&](size_t bytes) -> void* {
        void* r = (void*)wptr;
        wptr += (bytes + 255) & ~(size_t)255;
        return r;
    };
    const size_t SPEC = (size_t)BATCH * FBIN * TFRM;           // 129014
    const size_t HN   = (size_t)BATCH * NBAND * FEATC * TFRM;  // 1991936
    float* sre    = (float*)alloc(SPEC * 4);
    float* sim    = (float*)alloc(SPEC * 4);
    float* h      = (float*)alloc(HN * 4);
    float* ybuf   = (float*)alloc(HN * 4);
    float* xg     = (float*)alloc((size_t)62 * TFRM * G4 * 4);  // == 502*31*1024
    float* hcat   = (float*)alloc((size_t)62 * TFRM * 512 * 4);
    float* projo  = (float*)alloc(HN * 4);
    float* xcomm  = (float*)alloc(HN * 4);
    _Float16* w16 = (_Float16*)alloc((size_t)G4 * HID * 2);     // largest packed weight
    float* m1     = (float*)alloc((size_t)502 * 256 * 4);
    float* m2     = (float*)alloc((size_t)502 * 256 * 4);
    float* m3     = (float*)alloc((size_t)502 * 64 * 4);
    float* ere    = (float*)alloc(SPEC * 4);
    float* eim    = (float*)alloc(SPEC * 4);
    float* padded = (float*)alloc((size_t)BATCH * PADLEN * 4);

    auto cgrid = [](int n) { return (n + 255) / 256; };
    auto gemm = [&](const float* A, const float* W, const float* b1, const float* b2,
                    float* C, int M, int N, int K, int act) {
        int n16pad = ((N + 63) >> 6) * 4;                       // whole 64-col super-tiles
        int total = n16pad * (K >> 5) * 512;
        pack_frag_kernel<<<cgrid(total), 256, 0, stream>>>(W, w16, N, K, n16pad);
        dim3 g((N + 63) / 64, (M + 63) / 64);
        wmma_gemm_kernel<<<g, 128, 0, stream>>>(A, w16, b1, b2, C, M, N, K, act);
    };

    // ---- STFT ----
    stft_kernel<<<BATCH * TFRM, 256, 0, stream>>>(noisy, sre, sim);

    // ---- band split ----
    for (int i = 0; i < NBAND; ++i)
        bandsplit_kernel<<<BATCH, 256, 0, stream>>>(sre, sim, F0[i], BW[i], i,
                                                    bn[i].gnw, bn[i].gnb, bn[i].cw, bn[i].cb, h);

    // ---- res_rnn helper ----
    auto run_res = [&](const RESP& q, float* x, int S, int Tq) {
        int C = FEATC, M = S * Tq;
        gn_transpose_kernel<<<S, 256, 0, stream>>>(x, (size_t)C * Tq, C, Tq, q.gnw, q.gnb, ybuf);
        // forward LSTM
        gemm(ybuf, q.wihf, q.bihf, q.bhhf, xg, M, G4, C, 0);
        pack_frag_kernel<<<cgrid(64 * 8 * 512), 256, 0, stream>>>(q.whhf, w16, G4, HID, 64);
        lstm_scan_kernel<<<(S + 15) / 16, 128, 0, stream>>>(xg, w16, hcat, S, Tq, 0, 0);
        // backward LSTM
        gemm(ybuf, q.wihb, q.bihb, q.bhhb, xg, M, G4, C, 0);
        pack_frag_kernel<<<cgrid(64 * 8 * 512), 256, 0, stream>>>(q.whhb, w16, G4, HID, 64);
        lstm_scan_kernel<<<(S + 15) / 16, 128, 0, stream>>>(xg, w16, hcat, S, Tq, HID, 1);
        // projection + residual
        gemm(hcat, q.projw, q.projb, nullptr, projo, M, C, 512, 0);
        add_transpose_kernel<<<cgrid(S * C * Tq), 256, 0, stream>>>(x, projo, S, C, Tq);
    };

    // ---- separator: 4 repeats of {band_rnn over time, band_comm over bands} ----
    for (int rep = 0; rep < 4; ++rep) {
        run_res(sep[rep][0], h, BATCH * NBAND, TFRM);                       // S=62,  Tq=251
        gather_comm_kernel<<<cgrid((int)HN), 256, 0, stream>>>(h, xcomm);
        run_res(sep[rep][1], xcomm, BATCH * TFRM, NBAND);                   // S=502, Tq=31
        scatter_comm_kernel<<<cgrid((int)HN), 256, 0, stream>>>(h, xcomm);
    }

    // ---- mask estimation + complex masking ----
    for (int i = 0; i < NBAND; ++i) {
        const float* base = h + (size_t)i * FEATC * TFRM;                   // sep[:, i]
        gn_transpose_kernel<<<BATCH, 256, 0, stream>>>(base, (size_t)NBAND * FEATC * TFRM,
                                                       FEATC, TFRM, mk[i].gnw, mk[i].gnb, ybuf);
        int M = BATCH * TFRM;
        gemm(ybuf, mk[i].w1, mk[i].b1, nullptr, m1, M, 256, 128, 1);
        gemm(m1,   mk[i].w2, mk[i].b2, nullptr, m2, M, 256, 256, 1);
        gemm(m2,   mk[i].w3, mk[i].b3, nullptr, m3, M, 4 * BW[i], 256, 0);
        mask_apply_kernel<<<cgrid(BATCH * BW[i] * TFRM), 256, 0, stream>>>(
            m3, sre, sim, F0[i], BW[i], ere, eim);
    }

    // ---- iSTFT ----
    zero_kernel<<<cgrid(BATCH * PADLEN), 256, 0, stream>>>(padded, BATCH * PADLEN);
    istft_kernel<<<BATCH * TFRM, 256, 0, stream>>>(ere, eim, padded);
    istft_norm_kernel<<<cgrid(BATCH * LSAMP), 256, 0, stream>>>(padded, (float*)d_out);
}
